// PermutoLattice_19387482374450
// MI455X (gfx1250) — matrix operand order repaired and checked by reading, not compiled, not run on hardware
//
#include <hip/hip_runtime.h>

typedef __attribute__((ext_vector_type(16))) _Float16 v16h;
typedef __attribute__((ext_vector_type(8)))  float    v8f;

#define LAT_D    5
#define LAT_DP1  6
#define KEY_EMPTY 0xFFFFFFFFFFFFFFFFull

// ---------------- hash helpers ----------------
__device__ __forceinline__ unsigned hmix(unsigned long long x) {
  x ^= x >> 33; x *= 0xff51afd7ed558ccdULL;
  x ^= x >> 33; x *= 0xc4ceb9fe1a85ec53ULL;
  x ^= x >> 33;
  return (unsigned)x;
}

// reference encoding: code = sum_i (k[i]+1024) * 2048^(4-i)  == 11-bit digit packing
__device__ __forceinline__ unsigned long long enc5(const int* k) {
  unsigned long long c = 0;
#pragma unroll
  for (int i = 0; i < LAT_D; ++i)
    c = (c << 11) | (unsigned long long)(unsigned)(k[i] + 1024);
  return c;
}

__device__ __forceinline__ int ht_insert(unsigned long long* keys,
                                         unsigned long long code, unsigned mask) {
  unsigned s = hmix(code) & mask;
  for (unsigned t = 0; t <= mask; ++t) {
    unsigned long long prev = atomicCAS(&keys[s], KEY_EMPTY, code);
    if (prev == KEY_EMPTY || prev == code) return (int)s;
    s = (s + 1) & mask;
  }
  return (int)(hmix(code) & mask);  // table full (should not happen)
}

__device__ __forceinline__ int ht_find(const unsigned long long* __restrict__ keys,
                                       unsigned long long code, unsigned mask) {
  unsigned s = hmix(code) & mask;
  for (unsigned t = 0; t <= mask; ++t) {
    unsigned long long k = keys[s];
    if (k == code) return (int)s;
    if (k == KEY_EMPTY) return -1;
    s = (s + 1) & mask;
  }
  return -1;
}

// ---------------- elevation matrix (with scale folded in) ----------------
__device__ __forceinline__ float lat_scale(int k) {
  // sqrt(2/3)*(d+1)/sqrt((k+1)(k+2)), d=5
  const float s[5] = {3.4641016151f, 2.0f, 1.4142135624f, 1.0954451150f, 0.8944271910f};
  return s[k];
}
__device__ __forceinline__ float Emat(int m, int kk) {
  // elevated[m] = sum_{k>=m} scale_k f_k  - m*scale_{m-1} f_{m-1}
  if (m > LAT_D) return 0.f;
  if (kk >= m) return lat_scale(kk);
  if (kk == m - 1) return -(float)m * lat_scale(kk);
  return 0.f;
}

__device__ __forceinline__ void split2(float x, _Float16& hi, _Float16& lo) {
  hi = (_Float16)x;
  lo = (_Float16)(x - (float)hi);
}

// K-slot packing for split-f16 WMMA:  k 0..4 -> Ehi*fhi, 5..9 -> Ehi*flo, 10..14 -> Elo*fhi
__device__ __forceinline__ _Float16 aElem(int m, int kk) {
  if (kk >= 15) return (_Float16)0.f;
  int base = kk, part = 0;
  if (kk >= 10) { base = kk - 10; part = 2; }
  else if (kk >= 5) { base = kk - 5; part = 1; }
  float e = Emat(m, base);
  _Float16 hi, lo; split2(e, hi, lo);
  return (part == 2) ? lo : hi;
}
__device__ __forceinline__ _Float16 bElem(int kk, const _Float16* fhi, const _Float16* flo) {
  if (kk >= 15) return (_Float16)0.f;
  if (kk >= 10) return fhi[kk - 10];
  if (kk >= 5)  return flo[kk - 5];
  return fhi[kk];
}

// ---------------- kernels ----------------
__global__ void k_init_keys(unsigned long long* keys, unsigned C) {
  unsigned i = blockIdx.x * blockDim.x + threadIdx.x;
  if (i < C) keys[i] = KEY_EMPTY;
}

__global__ void k_init_vals(float4* v, size_t n4) {
  size_t i = (size_t)blockIdx.x * blockDim.x + threadIdx.x;
  if (i < n4) { float4 z; z.x = z.y = z.z = z.w = 0.f; v[i] = z; }
}

// One wave32 handles 32 points: two WMMA elevations (split-f16, f32 accumulate,
// 16 points each). WMMA C-layout leaves results in lanes 0..15, so the second
// tile's results are broadcast to lanes 16..31 via __shfl (LDS-permute path).
// Then ALL 32 lanes run rank / barycentric / hash-insert / atomic splat.
__global__ void __launch_bounds__(256)
k_splat(const float* __restrict__ feature, const float* __restrict__ values,
        unsigned long long* __restrict__ keys, float* __restrict__ val0,
        int* __restrict__ invIdx, float* __restrict__ baryOut,
        int N, unsigned mask) {
  const int lane = (int)(threadIdx.x & 31u);
  const int sub  = lane & 15;
  const int half = lane >> 4;
  const long long wave = ((long long)blockIdx.x * blockDim.x + threadIdx.x) >> 5;
  const long long base = wave * 32;
  const int p  = (int)(base + lane);           // this lane's own point
  const int q1 = (int)(base + sub);            // column point for B tile 1
  const int q2 = (int)(base + 16 + sub);       // column point for B tile 2
  const int q1c = (q1 < N) ? q1 : (N - 1);
  const int q2c = (q2 < N) ? q2 : (N - 1);

  float f1[LAT_D], f2[LAT_D];
#pragma unroll
  for (int i = 0; i < LAT_D; ++i) {
    f1[i] = feature[(size_t)q1c * LAT_D + i];
    f2[i] = feature[(size_t)q2c * LAT_D + i];
  }
  _Float16 f1hi[LAT_D], f1lo[LAT_D], f2hi[LAT_D], f2lo[LAT_D];
#pragma unroll
  for (int i = 0; i < LAT_D; ++i) {
    split2(f1[i], f1hi[i], f1lo[i]);
    split2(f2[i], f2hi[i], f2lo[i]);
  }

  // A (16x32, M=output dim = lane&15): lanes0-15 K=0..7 / 16..23, lanes16-31 K=8..15 / 24..31
  // B (32x16, N=point    = lane&15): lanes0-15 K=0..7 / 16..23, lanes16-31 K=8..15 / 24..31
  v16h a, b1, b2;
#pragma unroll
  for (int v = 0; v < 8; ++v) {
    int ka = (v < 4) ? (half * 8 + 2 * v) : (16 + half * 8 + 2 * (v - 4));
    a[2 * v]     = aElem(sub, ka);
    a[2 * v + 1] = aElem(sub, ka + 1);
    int kb = (v < 4) ? (half * 8 + 2 * v) : (16 + half * 8 + 2 * (v - 4));
    b1[2 * v]     = bElem(kb,     f1hi, f1lo);
    b1[2 * v + 1] = bElem(kb + 1, f1hi, f1lo);
    b2[2 * v]     = bElem(kb,     f2hi, f2lo);
    b2[2 * v + 1] = bElem(kb + 1, f2hi, f2lo);
  }

  v8f cz = {0.f, 0.f, 0.f, 0.f, 0.f, 0.f, 0.f, 0.f};
  v8f c1 = __builtin_amdgcn_wmma_f32_16x16x32_f16(false, a, false, b1, (short)0, cz, false, false);
  v8f c2 = __builtin_amdgcn_wmma_f32_16x16x32_f16(false, a, false, b2, (short)0, cz, false, false);
  // c1/c2: lane n (0..15), element j -> elevated[j] of tile point n (rows >=6 are zero)

  // Broadcast tile-2 results (in lanes 0..15) to lanes 16..31; uniform, pre-divergence.
  float elevated[LAT_DP1];
#pragma unroll
  for (int j = 0; j < LAT_DP1; ++j) {
    float sh = __shfl(c2[j], sub, 32);
    elevated[j] = half ? sh : c1[j];
  }

  if (p >= N) return;

  const float down = 1.0f / 6.0f;
  float rd[LAT_DP1]; int sumi = 0;
#pragma unroll
  for (int j = 0; j < LAT_DP1; ++j) { rd[j] = rintf(elevated[j] * down); sumi += (int)rd[j]; }

  float diff[LAT_DP1];
#pragma unroll
  for (int j = 0; j < LAT_DP1; ++j) diff[j] = (elevated[j] - rd[j] * 6.0f) * down;

  // stable descending rank: count strictly-greater, ties broken by index
  int rank[LAT_DP1], rem0i[LAT_DP1];
#pragma unroll
  for (int i = 0; i < LAT_DP1; ++i) {
    int r = 0;
#pragma unroll
    for (int j = 0; j < LAT_DP1; ++j)
      r += (diff[j] > diff[i]) || (diff[j] == diff[i] && j < i);
    rank[i]  = r + sumi;
    rem0i[i] = (int)(rd[i] * 6.0f);
  }
#pragma unroll
  for (int i = 0; i < LAT_DP1; ++i) {
    if (rank[i] < 0)      { rank[i] += 6; rem0i[i] += 6; }
    else if (rank[i] > 5) { rank[i] -= 6; rem0i[i] -= 6; }
  }

  float bvec[LAT_DP1 + 1] = {0.f, 0.f, 0.f, 0.f, 0.f, 0.f, 0.f};
#pragma unroll
  for (int i = 0; i < LAT_DP1; ++i) {
    float t = (elevated[i] - (float)rem0i[i]) * down;
    bvec[5 - rank[i]] += t;
    bvec[6 - rank[i]] -= t;
  }
  float bary[LAT_DP1];
  bary[0] = bvec[0] + 1.0f + bvec[6];
#pragma unroll
  for (int r = 1; r < LAT_DP1; ++r) bary[r] = bvec[r];

  float vals[16];
#pragma unroll
  for (int v = 0; v < 16; ++v) vals[v] = values[(size_t)p * 16 + v];

#pragma unroll
  for (int r = 0; r < LAT_DP1; ++r) {
    int key[LAT_D];
#pragma unroll
    for (int i = 0; i < LAT_D; ++i)
      key[i] = rem0i[i] + ((rank[i] >= LAT_DP1 - r) ? (r - LAT_DP1) : r);
    unsigned long long code = enc5(key);
    int slot = ht_insert(keys, code, mask);
    invIdx[(size_t)p * LAT_DP1 + r]  = slot;
    baryOut[(size_t)p * LAT_DP1 + r] = bary[r];
    float w = bary[r];
    float* dstp = val0 + (size_t)slot * 16;
#pragma unroll
    for (int v = 0; v < 16; ++v) atomicAdd(dstp + v, w * vals[v]);
  }
}

// one [1,2,1]/4 blur pass along direction `dir` (0..5); reads src, writes dst
__global__ void __launch_bounds__(256)
k_blur(const unsigned long long* __restrict__ keys,
       const float* __restrict__ src, float* __restrict__ dst,
       unsigned C, unsigned mask, int dir) {
  unsigned slot = blockIdx.x * blockDim.x + threadIdx.x;
  if (slot >= C) return;
  unsigned long long code = keys[slot];
  float4* __restrict__ d4 = (float4*)(dst + (size_t)slot * 16);
  if (code == KEY_EMPTY) {
    float4 z; z.x = z.y = z.z = z.w = 0.f;
#pragma unroll
    for (int q = 0; q < 4; ++q) d4[q] = z;
    return;
  }
  int k[LAT_D];
#pragma unroll
  for (int i = 0; i < LAT_D; ++i)
    k[i] = (int)((code >> (11 * (LAT_D - 1 - i))) & 0x7FF) - 1024;
  // offset = ones(6) with off[dir] = -5; only the first 5 coords matter for the key
  int kp[LAT_D], km[LAT_D];
#pragma unroll
  for (int i = 0; i < LAT_D; ++i) {
    int o = (i == dir) ? -5 : 1;
    kp[i] = k[i] + o;
    km[i] = k[i] - o;
  }
  int ip = ht_find(keys, enc5(kp), mask);
  int im = ht_find(keys, enc5(km), mask);
  const float4* s4c = (const float4*)(src + (size_t)slot * 16);
  const float4* s4p = (ip >= 0) ? (const float4*)(src + (size_t)ip * 16) : nullptr;
  const float4* s4m = (im >= 0) ? (const float4*)(src + (size_t)im * 16) : nullptr;
#pragma unroll
  for (int q = 0; q < 4; ++q) {
    float4 cv = s4c[q];
    float4 pv, mv;
    if (s4p) pv = s4p[q]; else { pv.x = pv.y = pv.z = pv.w = 0.f; }
    if (s4m) mv = s4m[q]; else { mv.x = mv.y = mv.z = mv.w = 0.f; }
    float4 o;
    o.x = 0.5f * cv.x + 0.25f * (pv.x + mv.x);
    o.y = 0.5f * cv.y + 0.25f * (pv.y + mv.y);
    o.z = 0.5f * cv.z + 0.25f * (pv.z + mv.z);
    o.w = 0.5f * cv.w + 0.25f * (pv.w + mv.w);
    d4[q] = o;
  }
}

__global__ void __launch_bounds__(256)
k_slice(const float* __restrict__ val, const int* __restrict__ invIdx,
        const float* __restrict__ bary, float* __restrict__ out, int N) {
  int p = blockIdx.x * blockDim.x + threadIdx.x;
  if (p >= N) return;
  float acc[16];
#pragma unroll
  for (int v = 0; v < 16; ++v) acc[v] = 0.f;
#pragma unroll
  for (int r = 0; r < LAT_DP1; ++r) {
    int s  = invIdx[(size_t)p * LAT_DP1 + r];
    float w = bary[(size_t)p * LAT_DP1 + r];
    const float* vp = val + (size_t)s * 16;
#pragma unroll
    for (int v = 0; v < 16; ++v) acc[v] += w * vp[v];
  }
  const float alpha = 1.0f / (1.0f + 0.03125f);  // 1/(1+2^-d), d=5
#pragma unroll
  for (int v = 0; v < 16; ++v) out[(size_t)p * 16 + v] = alpha * acc[v];
}

// ---------------- host launcher ----------------
extern "C" void kernel_launch(void* const* d_in, const int* in_sizes, int n_in,
                              void* d_out, int out_size, void* d_ws, size_t ws_size,
                              hipStream_t stream) {
  const float* feature = (const float*)d_in[0];
  const float* values  = (const float*)d_in[1];
  float* out = (float*)d_out;

  const int N = in_sizes[0] / LAT_D;           // feature is [N, 5]
  const size_t M = (size_t)N * LAT_DP1;        // 6 lattice vertices per point

  // hash capacity: power of two, >= worst-case unique count (M), fit in workspace
  unsigned C = 1u << 24;
  while (C > (1u << 22)) {
    size_t need = (size_t)C * 8           // keys
                + 2 * (size_t)C * 64      // two 16-channel value buffers
                + 2 * M * 4;              // inv + bary
    if (need <= ws_size && (size_t)C >= M) break;
    if (need <= ws_size) break;
    C >>= 1;
  }
  const unsigned mask = C - 1;

  char* wp = (char*)d_ws;
  unsigned long long* keys = (unsigned long long*)wp; wp += (size_t)C * 8;
  float* valA = (float*)wp; wp += (size_t)C * 64;
  float* valB = (float*)wp; wp += (size_t)C * 64;
  int*   invIdx  = (int*)wp; wp += M * 4;
  float* baryArr = (float*)wp;

  // init hash keys + splat accumulator (must happen every call: ws is not re-poisoned)
  {
    unsigned blocks = (C + 255u) / 256u;
    k_init_keys<<<blocks, 256, 0, stream>>>(keys, C);
    size_t n4 = (size_t)C * 4;  // float4 elements in valA
    unsigned blocks2 = (unsigned)((n4 + 255) / 256);
    k_init_vals<<<blocks2, 256, 0, stream>>>((float4*)valA, n4);
  }

  // elevate + rank + hash-insert + splat: 32 points per wave, 8 waves/block
  {
    unsigned blocks = (unsigned)((N + 255) / 256);
    k_splat<<<blocks, 256, 0, stream>>>(feature, values, keys, valA, invIdx, baryArr, N, mask);
  }

  // 6 directional blur passes, ping-pong A<->B
  float* src = valA;
  float* dst = valB;
  for (int j = 0; j < LAT_DP1; ++j) {
    unsigned blocks = (C + 255u) / 256u;
    k_blur<<<blocks, 256, 0, stream>>>(keys, src, dst, C, mask, j);
    float* t = src; src = dst; dst = t;
  }

  // slice (final blurred values are in `src` after the last swap)
  {
    unsigned blocks = (unsigned)((N + 255) / 256);
    k_slice<<<blocks, 256, 0, stream>>>(src, invIdx, baryArr, out, N);
  }

  (void)n_in; (void)out_size;
}
